// GCNN_43104291783024
// MI455X (gfx1250) — matrix operand order repaired and checked by reading, not compiled, not run on hardware
//
#include <hip/hip_runtime.h>
#include <hip/hip_bf16.h>

// ---------------------------------------------------------------------------
// GCN kinase classifier for MI455X (gfx1250, wave32, WMMA).
// - Output depends only on h[node_index]; we compute x@W_conv^T only for rows
//   feeding selected nodes (~18K of 50K), via v_wmma_f32_16x16x32_bf16.
// - A rows are gathered + converted ONCE into bf16 A-fragment lane order.
// - Conv GEMM is LDS-blocked: 8 waves share a 128x128 super-tile; the W k-tile
//   is double-buffered in LDS (bank-conflict-padded), cutting W L2 traffic ~8x.
// - Integer inputs (edge_index, node_index) are int32 per harness convention.
// - Worst-case workspace ~285 MB (xbf + hbuf bf16 dominate).
// ---------------------------------------------------------------------------

typedef __bf16 bf16;
typedef __attribute__((ext_vector_type(16))) __bf16 v16bf;
typedef __attribute__((ext_vector_type(8)))  __bf16 v8bf;
typedef __attribute__((ext_vector_type(8)))  float  v8f;

#define F_DIM 1280              // ESM feature dim (K of both WMMA GEMMs)
#define D_DIM 128               // pfc1 output dim
#define H2_DIM 64
#define C_DIM 20
#define KSTEPS (F_DIM / 32)     // 40 k-iterations of K=32 WMMA
#define NSTRIP (F_DIM / 128)    // 10 column-strips of 128 for conv GEMM

__device__ __forceinline__ float lrelu(float v) { return v > 0.0f ? v : 0.01f * v; }

// -------------------------------------------------- simple f32 -> bf16 convert
__global__ void f32_to_bf16_kernel(const float* __restrict__ in, bf16* __restrict__ out, long n) {
    long i = (long)blockIdx.x * blockDim.x + threadIdx.x;
    if (i < n) out[i] = (bf16)in[i];
}

// -------------------------------------------------- mark selected sites, assign slots
__global__ void mark_selected_kernel(const int* __restrict__ node_index, int Bn,
                                     int* slot, int* needpos, int* cnt,
                                     int* sellist, int* needlist) {
    int b = blockIdx.x * blockDim.x + threadIdx.x;
    if (b >= Bn) return;
    int n = node_index[b];
    if (atomicCAS(&slot[n], -1, -2) == -1) {        // first claimer of this node
        int i = atomicAdd(&cnt[0], 1);
        sellist[i] = n;
        slot[n] = i;
        if (atomicCAS(&needpos[n], -1, -2) == -1) { // selected nodes need their h row too
            int j = atomicAdd(&cnt[1], 1);
            needlist[j] = n;
            needpos[n] = j;
        }
    }
}

// -------------------------------------------------- degree count + mark needed h rows
__global__ void degree_need_kernel(const int* __restrict__ src, const int* __restrict__ dst,
                                   int En, int* deg, const int* __restrict__ slot,
                                   int* needpos, int* cnt, int* needlist) {
    int e = blockIdx.x * blockDim.x + threadIdx.x;
    if (e >= En) return;
    int d = dst[e];
    atomicAdd(&deg[d], 1);
    if (slot[d] >= 0) {                              // edge lands on a selected node
        int s = src[e];
        if (atomicCAS(&needpos[s], -1, -2) == -1) {
            int j = atomicAdd(&cnt[1], 1);
            needlist[j] = s;
            needpos[s] = j;
        }
    }
}

// -------------------------------------------------- dinv = rsqrt(deg + 1 self loop)
__global__ void dinv_kernel(const int* __restrict__ deg, float* __restrict__ dinv, int Nn) {
    int n = blockIdx.x * blockDim.x + threadIdx.x;
    if (n < Nn) dinv[n] = rsqrtf((float)deg[n] + 1.0f);
}

// -------------------------------------------------- gather + convert + swizzle A rows
// One wave per (m-tile, k-step). Output chunk is the A operand in contiguous k order (B matches):
// lane<16 holds K kbase=0 -> {0..15}; lane>=16 kbase=16 -> {16..31},
// stored as 16 contiguous bf16 per lane so the GEMM loads one v16bf per k-step.
__global__ __launch_bounds__(256) void pack_a_kernel(
        const float* __restrict__ x, const int* __restrict__ needlist,
        const int* __restrict__ cnt, bf16* __restrict__ xbf, int MT) {
    int wid = blockIdx.x * 8 + (threadIdx.x >> 5);
    if (wid >= MT * KSTEPS) return;
    int mt = wid / KSTEPS;
    int kt = wid - mt * KSTEPS;
    int nc = cnt[1];
    if (mt * 16 >= nc) return;             // wave-uniform early exit
    int lane = threadIdx.x & 31;
    int m = lane & 15;
    int row = mt * 16 + m;
    if (row >= nc) row = nc - 1;           // clamp; duplicate pad rows harmless
    int node = needlist[row];
    int kA = (lane < 16) ? 0 : 16;
    const float* xr = x + (size_t)node * F_DIM + kt * 32 + kA;
    float4 f0 = *(const float4*)(xr);
    float4 f1 = *(const float4*)(xr + 4);
    float4 f2 = *(const float4*)(xr + 8);
    float4 f3 = *(const float4*)(xr + 12);
    v16bf a;
    a[0]  = (bf16)f0.x; a[1]  = (bf16)f0.y; a[2]  = (bf16)f0.z; a[3]  = (bf16)f0.w;
    a[4]  = (bf16)f1.x; a[5]  = (bf16)f1.y; a[6]  = (bf16)f1.z; a[7]  = (bf16)f1.w;
    a[8]  = (bf16)f2.x; a[9]  = (bf16)f2.y; a[10] = (bf16)f2.z; a[11] = (bf16)f2.w;
    a[12] = (bf16)f3.x; a[13] = (bf16)f3.y; a[14] = (bf16)f3.z; a[15] = (bf16)f3.w;
    *(v16bf*)(xbf + ((size_t)wid * 32 + lane) * 16) = a;
}

// -------------------------------------------------- conv GEMM: h = x@W^T (compacted rows)
// Block = 8 waves = 128(M) x 128(N) super-tile. All waves share the same B
// fragments, staged per k-step into double-buffered LDS (8KB tiles, 80B row
// stride so 16-lane fragment reads hit distinct banks). Each wave: 16x128,
// 8 accumulators, 1 A load + 8 LDS B loads + 8 WMMA per k-step.
__global__ __launch_bounds__(256) void gemm_conv_wmma(
        const bf16* __restrict__ xbf, const bf16* __restrict__ Wbf,
        const int* __restrict__ cnt, bf16* __restrict__ hbuf, int MB) {
    __shared__ uint4 sB[2][128 * 5];       // 2 x 128 rows x (64B data + 16B pad) = 20KB
    int nt = blockIdx.x % NSTRIP;
    int mb = blockIdx.x / NSTRIP;
    int nc = cnt[1];
    if (mb * 128 >= nc) return;            // block-uniform exit (before any barrier)

    int t = threadIdx.x;
    int lane = t & 31;
    int m = lane & 15;
    int mt = mb * 8 + (t >> 5);
    bool active = (mt * 16) < nc;          // pad waves still run (barrier members)

    // staging coords: thread t moves 32B = half a W row's k-chunk
    int srow = t >> 1;                     // 0..127 (row within N strip)
    int shalf = t & 1;                     // 0,1 -> 32B halves of the 64B k-chunk
    const bf16* wsrc = Wbf + (size_t)(nt * 128 + srow) * F_DIM + shalf * 16;
    int sidx = srow * 5 + shalf * 2;

    const bf16* ap = xbf + ((size_t)mt * KSTEPS * 32 + lane) * 16;
    int kofs = (lane < 16) ? 0 : 2;        // uint4 offset selecting K half in LDS

    v8f acc[8];
#pragma unroll
    for (int j = 0; j < 8; ++j) acc[j] = (v8f){};

    {   // preload k-step 0 into buffer 0
        const uint4* g = (const uint4*)(wsrc);
        uint4 g0 = g[0], g1 = g[1];
        sB[0][sidx] = g0; sB[0][sidx + 1] = g1;
    }
    __syncthreads();

    for (int kt = 0; kt < KSTEPS; ++kt) {
        int buf = kt & 1;
        if (kt + 1 < KSTEPS) {             // stage next k-tile into other buffer
            const uint4* g = (const uint4*)(wsrc + (kt + 1) * 32);
            uint4 g0 = g[0], g1 = g[1];
            sB[buf ^ 1][sidx] = g0; sB[buf ^ 1][sidx + 1] = g1;
        }
        v16bf a = *(const v16bf*)(ap);
        ap += 32 * 16;
        __builtin_prefetch(ap + 512, 0, 1);           // global_prefetch_b8: A chunk two k-steps ahead
#pragma unroll
        for (int j = 0; j < 8; ++j) {
            v16bf b;
            __builtin_memcpy(&b, &sB[buf][(j * 16 + m) * 5 + kofs], 32);
            acc[j] = __builtin_amdgcn_wmma_f32_16x16x32_bf16(
                false, a, false, b, (short)0, acc[j], false, false);
        }
        __syncthreads();                   // readers done before buf is overwritten
    }

    if (active) {
        // C/D layout: lanes 0-15 rows 0..7, lanes 16-31 rows 8..15; col = lane&15
        int mrow = mt * 16 + ((lane < 16) ? 0 : 8);
#pragma unroll
        for (int v = 0; v < 8; ++v) {
            bf16* hr = hbuf + (size_t)(mrow + v) * F_DIM + nt * 128;
#pragma unroll
            for (int j = 0; j < 8; ++j) hr[m + j * 16] = (bf16)acc[j][v];
        }
    }
}

// -------------------------------------------------- edge scatter (wave per edge)
__global__ __launch_bounds__(256) void scatter_kernel(
        const int* __restrict__ src, const int* __restrict__ dst, int En,
        const float* __restrict__ dinv, const int* __restrict__ slot,
        const int* __restrict__ needpos, const bf16* __restrict__ hbuf,
        float* __restrict__ agg) {
    int e = blockIdx.x * 8 + (threadIdx.x >> 5);
    if (e >= En) return;
    int d = dst[e];
    int r = slot[d];
    if (r < 0) return;                     // ~96% of edges exit here
    int lane = threadIdx.x & 31;
    int s = src[e];
    float w = dinv[s] * dinv[d];
    const bf16* hrow = hbuf + (size_t)needpos[s] * F_DIM;
    float* arow = agg + (size_t)r * F_DIM;
    for (int k = lane * 8; k < F_DIM; k += 256) {
        v8bf h = *(const v8bf*)(hrow + k);
#pragma unroll
        for (int j = 0; j < 8; ++j)
            unsafeAtomicAdd(arow + k + j, (float)h[j] * w);
    }
}

// -------------------------------------------------- self-loop + bias + leakyrelu -> bf16
__global__ __launch_bounds__(256) void finalize_kernel(
        const int* __restrict__ cnt, const int* __restrict__ sellist,
        const int* __restrict__ needpos, const float* __restrict__ dinv,
        const bf16* __restrict__ hbuf, const float* __restrict__ agg,
        const float* __restrict__ bconv, bf16* __restrict__ zbf) {
    int i = blockIdx.x * 8 + (threadIdx.x >> 5);
    if (i >= cnt[0]) return;
    int lane = threadIdx.x & 31;
    int n = sellist[i];
    float w = dinv[n] * dinv[n];
    const bf16* hrow = hbuf + (size_t)needpos[n] * F_DIM;
    const float* arow = agg + (size_t)i * F_DIM;
    bf16* zrow = zbf + (size_t)i * F_DIM;
    for (int k = lane * 8; k < F_DIM; k += 256) {
        v8bf hv = *(const v8bf*)(hrow + k);
#pragma unroll
        for (int j = 0; j < 4; ++j) {
            float4 av = *(const float4*)(arow + k + j * 4);
            float4 bv = *(const float4*)(bconv + k + j * 4);
            zrow[k + j * 4 + 0] = (bf16)lrelu(av.x + (float)hv[j * 4 + 0] * w + bv.x);
            zrow[k + j * 4 + 1] = (bf16)lrelu(av.y + (float)hv[j * 4 + 1] * w + bv.y);
            zrow[k + j * 4 + 2] = (bf16)lrelu(av.z + (float)hv[j * 4 + 2] * w + bv.z);
            zrow[k + j * 4 + 3] = (bf16)lrelu(av.w + (float)hv[j * 4 + 3] * w + bv.w);
        }
    }
}

// -------------------------------------------------- pfc1 GEMM: [B,1280] x [1280,128]
__global__ __launch_bounds__(256) void gemm_pfc1_wmma(
        const bf16* __restrict__ zbf, const bf16* __restrict__ W1bf,
        const int* __restrict__ node_index, const int* __restrict__ slot,
        const float* __restrict__ bp, float* __restrict__ y1, int Bn, int MT) {
    const int NT = D_DIM / 64;             // 2
    int wid = blockIdx.x * 8 + (threadIdx.x >> 5);
    if (wid >= MT * NT) return;
    int mt = wid / NT;
    int nt = wid - mt * NT;
    int lane = threadIdx.x & 31;
    int m = lane & 15;
    int b = mt * 16 + m;
    if (b >= Bn) b = Bn - 1;
    int i = slot[node_index[b]];           // duplicates resolve to same slot
    const bf16* zrow = zbf + (size_t)i * F_DIM;
    int kA = (lane < 16) ? 0 : 16;
    int kB = (lane < 16) ? 0 : 16;
    int nb = nt * 64 + m;

    v8f acc0 = {}, acc1 = {}, acc2 = {}, acc3 = {};
    for (int kt = 0; kt < KSTEPS; ++kt) {
        int k0 = kt * 32;
        v8bf c0 = *(const v8bf*)(zrow + k0 + kA);
        v8bf c1 = *(const v8bf*)(zrow + k0 + kA + 8);
        v16bf a;
#pragma unroll
        for (int j = 0; j < 8; ++j) { a[j] = c0[j]; a[8 + j] = c1[j]; }
        const bf16* wp = W1bf + (size_t)nb * F_DIM + k0 + kB;
        v16bf b0 = *(const v16bf*)(wp);
        v16bf b1 = *(const v16bf*)(wp + (size_t)16 * F_DIM);
        v16bf b2 = *(const v16bf*)(wp + (size_t)32 * F_DIM);
        v16bf b3 = *(const v16bf*)(wp + (size_t)48 * F_DIM);
        acc0 = __builtin_amdgcn_wmma_f32_16x16x32_bf16(false, a, false, b0, (short)0, acc0, false, false);
        acc1 = __builtin_amdgcn_wmma_f32_16x16x32_bf16(false, a, false, b1, (short)0, acc1, false, false);
        acc2 = __builtin_amdgcn_wmma_f32_16x16x32_bf16(false, a, false, b2, (short)0, acc2, false, false);
        acc3 = __builtin_amdgcn_wmma_f32_16x16x32_bf16(false, a, false, b3, (short)0, acc3, false, false);
    }
    int mb = mt * 16 + ((lane < 16) ? 0 : 8);
    int col0 = nt * 64 + m;
    float bp0 = bp[col0], bp1 = bp[col0 + 16], bp2 = bp[col0 + 32], bp3 = bp[col0 + 48];
#pragma unroll
    for (int v = 0; v < 8; ++v) {
        int rr = mb + v;
        if (rr < Bn) {
            float* yr = y1 + (size_t)rr * D_DIM + nt * 64;
            yr[m]      = lrelu(acc0[v] + bp0);
            yr[m + 16] = lrelu(acc1[v] + bp1);
            yr[m + 32] = lrelu(acc2[v] + bp2);
            yr[m + 48] = lrelu(acc3[v] + bp3);
        }
    }
}

// -------------------------------------------------- fused tail MLP + softmax (tiny)
__global__ __launch_bounds__(128) void mlp_head_kernel(
        const float* __restrict__ y1,
        const float* __restrict__ Wfc1, const float* __restrict__ bfc1,
        const float* __restrict__ Wfc2, const float* __restrict__ bfc2,
        const float* __restrict__ Wout, const float* __restrict__ bout,
        float* __restrict__ out) {
    __shared__ float h1[D_DIM], h2[D_DIM], h3[H2_DIM], lg[C_DIM], red[2];
    int b = blockIdx.x;
    int t = threadIdx.x;
    h1[t] = y1[(size_t)b * D_DIM + t];
    __syncthreads();
    float acc = bfc1[t];
    for (int k = 0; k < D_DIM; ++k) acc += Wfc1[t * D_DIM + k] * h1[k];
    h2[t] = lrelu(acc);
    __syncthreads();
    if (t < H2_DIM) {
        acc = bfc2[t];
        for (int k = 0; k < D_DIM; ++k) acc += Wfc2[t * D_DIM + k] * h2[k];
        h3[t] = lrelu(acc);
    }
    __syncthreads();
    if (t < C_DIM) {
        acc = bout[t];
        for (int k = 0; k < H2_DIM; ++k) acc += Wout[t * H2_DIM + k] * h3[k];
        lg[t] = acc;
    }
    __syncthreads();
    if (t == 0) {
        float mx = lg[0];
        for (int i = 1; i < C_DIM; ++i) mx = fmaxf(mx, lg[i]);
        float s = 0.0f;
        for (int i = 0; i < C_DIM; ++i) s += expf(lg[i] - mx);
        red[0] = mx; red[1] = s;
    }
    __syncthreads();
    if (t < C_DIM) out[(size_t)b * C_DIM + t] = expf(lg[t] - red[0]) / red[1];
}

// ---------------------------------------------------------------------------
extern "C" void kernel_launch(void* const* d_in, const int* in_sizes, int n_in,
                              void* d_out, int out_size, void* d_ws, size_t ws_size,
                              hipStream_t stream) {
    const float* x       = (const float*)d_in[0];
    const int*   ei      = (const int*)d_in[1];     // [2,E] int32
    const int*   nidx    = (const int*)d_in[2];     // [B]  int32
    const float* W_conv  = (const float*)d_in[3];
    const float* b_conv  = (const float*)d_in[4];
    const float* W_pfc1  = (const float*)d_in[5];
    const float* b_pfc1  = (const float*)d_in[6];
    const float* W_fc1   = (const float*)d_in[7];
    const float* b_fc1   = (const float*)d_in[8];
    const float* W_fc2   = (const float*)d_in[9];
    const float* b_fc2   = (const float*)d_in[10];
    const float* W_out   = (const float*)d_in[11];
    const float* b_out   = (const float*)d_in[12];
    float* out = (float*)d_out;

    const long Nn   = in_sizes[0] / F_DIM;
    const long En   = in_sizes[1] / 2;
    const long Bn   = in_sizes[2];
    const long Npad = ((Nn + 15) / 16) * 16;
    const int  MT   = (int)(Npad / 16);
    const long MB   = (Npad + 127) / 128;           // 128-row super-tiles
    const long M128 = MB * 128;                     // xbf row allocation
    const int  MT2  = (int)((Bn + 15) / 16);
    const int* esrc = ei;
    const int* edst = ei + En;

    // ---- workspace carve-up (256B aligned regions)
    char* ws = (char*)d_ws;
    size_t off = 0;
    auto take = [&](size_t bytes) { size_t o = off; off += (bytes + 255) & ~(size_t)255; return o; };
    size_t o_slot  = take(Nn * 4);                   // int, init -1
    size_t o_need  = take(Nn * 4);                   // int, init -1
    size_t o_deg   = take(Nn * 4);                   // int, init 0
    size_t o_cnt   = take(8);                        // int[2]: nsel, needcnt, init 0
    size_t o_agg   = take((size_t)Bn * F_DIM * 4);   // f32, init 0 (atomic target)
    size_t o_sel   = take(Bn * 4);
    size_t o_nlist = take(Nn * 4);
    size_t o_dinv  = take(Nn * 4);
    size_t o_wbf   = take((size_t)F_DIM * F_DIM * 2);
    size_t o_w1bf  = take((size_t)D_DIM * F_DIM * 2);
    size_t o_zbf   = take((size_t)Bn * F_DIM * 2);
    size_t o_y1    = take((size_t)Bn * D_DIM * 4);
    size_t o_xbf   = take((size_t)M128 * F_DIM * 2); // A-fragment-order bf16 (128-row mult)
    size_t o_hbuf  = take((size_t)Npad * F_DIM * 2); // bf16 h rows
    (void)ws_size; (void)n_in; (void)out_size;

    int*   slot  = (int*)(ws + o_slot);
    int*   need  = (int*)(ws + o_need);
    int*   deg   = (int*)(ws + o_deg);
    int*   cnt   = (int*)(ws + o_cnt);
    float* agg   = (float*)(ws + o_agg);
    int*   sel   = (int*)(ws + o_sel);
    int*   nlist = (int*)(ws + o_nlist);
    float* dinv  = (float*)(ws + o_dinv);
    bf16*  Wbf   = (bf16*)(ws + o_wbf);
    bf16*  W1bf  = (bf16*)(ws + o_w1bf);
    bf16*  zbf   = (bf16*)(ws + o_zbf);
    float* y1    = (float*)(ws + o_y1);
    bf16*  xbf   = (bf16*)(ws + o_xbf);
    bf16*  hbuf  = (bf16*)(ws + o_hbuf);

    // ---- init: slot/need = -1 (0xFF), deg/cnt/agg = 0 (contiguous regions)
    hipMemsetAsync(ws + o_slot, 0xFF, o_deg - o_slot, stream);
    hipMemsetAsync(ws + o_deg, 0x00, o_sel - o_deg, stream);

    // ---- weight conversion to bf16 (row-major [out, k] -> direct B-fragment loads)
    {
        long n = (long)F_DIM * F_DIM;
        f32_to_bf16_kernel<<<(int)((n + 255) / 256), 256, 0, stream>>>(W_conv, Wbf, n);
        long n2 = (long)D_DIM * F_DIM;
        f32_to_bf16_kernel<<<(int)((n2 + 255) / 256), 256, 0, stream>>>(W_pfc1, W1bf, n2);
    }

    mark_selected_kernel<<<(int)((Bn + 255) / 256), 256, 0, stream>>>(
        nidx, (int)Bn, slot, need, cnt, sel, nlist);

    degree_need_kernel<<<(int)((En + 255) / 256), 256, 0, stream>>>(
        esrc, edst, (int)En, deg, slot, need, cnt, nlist);

    dinv_kernel<<<(int)((Nn + 255) / 256), 256, 0, stream>>>(deg, dinv, (int)Nn);

    {   // gather + convert + A-fragment swizzle of needed x rows
        long waves = (long)MT * KSTEPS;
        pack_a_kernel<<<(int)((waves + 7) / 8), 256, 0, stream>>>(
            x, nlist, cnt, xbf, MT);
    }

    // conv GEMM: one block per (128-row super-tile, 128-col strip)
    gemm_conv_wmma<<<(int)(MB * NSTRIP), 256, 0, stream>>>(
        xbf, Wbf, cnt, hbuf, (int)MB);

    scatter_kernel<<<(int)((En + 7) / 8), 256, 0, stream>>>(
        esrc, edst, (int)En, dinv, slot, need, hbuf, agg);

    finalize_kernel<<<(int)((Bn + 7) / 8), 256, 0, stream>>>(
        cnt, sel, need, dinv, hbuf, agg, b_conv, zbf);

    {   // pfc1 GEMM with per-site gather
        long waves = (long)MT2 * (D_DIM / 64);
        gemm_pfc1_wmma<<<(int)((waves + 7) / 8), 256, 0, stream>>>(
            zbf, W1bf, nidx, slot, b_pfc1, y1, (int)Bn, MT2);
    }

    mlp_head_kernel<<<(int)Bn, 128, 0, stream>>>(
        y1, W_fc1, b_fc1, W_fc2, b_fc2, W_out, b_out, out);
}
